// Net_68058051772416
// MI455X (gfx1250) — matrix operand order repaired and checked by reading, not compiled, not run on hardware
//
#include <hip/hip_runtime.h>
#include <hip/hip_bf16.h>
#include <math.h>

typedef __attribute__((ext_vector_type(16))) _Float16 v16h;
typedef __attribute__((ext_vector_type(8)))  _Float16 v8h;
typedef __attribute__((ext_vector_type(2)))  _Float16 v2h;
typedef __attribute__((ext_vector_type(8)))  float    v8f;

#define NOUT 512
#define WT_K 800            // 784 padded to 25 chunks of 32
#define AB_STRIDE 40        // halves; 80 B row stride, 16B aligned
#define C_STRIDE 516        // floats

// LDS layout (bytes)
#define OFF_AS   0                       // 2 x (32*40*2)  = 5120
#define OFF_BS   5120                    // 2 x (512*40*2) = 81920
#define OFF_BIAS 87040                   // 512*4         = 2048
#define OFF_CS   89088                   // 32*516*4      = 66048
#define SMEM_TOTAL (89088 + 32 * C_STRIDE * 4)   // 155136

// ---------------------------------------------------------------------------
// Kernel 1: collapse TT cores into dense transposed W: Wt[n=512][k=800] f16,
// zero-padded k rows 784..799. One block per output column c.
// ---------------------------------------------------------------------------
__global__ void tt_build_w(const float* __restrict__ c1,   // [7,4,20]
                           const float* __restrict__ c2,   // [20,4,4,20]
                           const float* __restrict__ c3,   // [20,7,8,16]
                           const float* __restrict__ c4,   // [16,4,4]
                           _Float16* __restrict__ Wt)      // [512][800]
{
    const int c = blockIdx.x;            // 0..511 : ((p*4+q)*8+e)*4+f
    const int f = c & 3;
    const int e = (c >> 2) & 7;
    const int q = (c >> 5) & 3;
    const int p = c >> 7;

    __shared__ float aS[7 * 4 * 20];          // a[i,j,s]
    __shared__ float btS[7 * 4 * 7 * 16];     // bt[i,j,k,t]

    // step 1: a[i,j,s] = sum_r c1[i,p,r] * c2[r,j,q,s]
    for (int idx = threadIdx.x; idx < 560; idx += blockDim.x) {
        int i = idx / 80;
        int j = (idx / 20) % 4;
        int s = idx % 20;
        float acc = 0.f;
        for (int r = 0; r < 20; ++r)
            acc += c1[(i * 4 + p) * 20 + r] * c2[((r * 4 + j) * 4 + q) * 20 + s];
        aS[(i * 4 + j) * 20 + s] = acc;
    }
    __syncthreads();

    // step 2: bt[i,j,k,t] = sum_s a[i,j,s] * c3[s,k,e,t]
    for (int idx = threadIdx.x; idx < 3136; idx += blockDim.x) {
        int t = idx & 15;
        int k = (idx >> 4) % 7;
        int j = (idx / 112) & 3;
        int i = idx / 448;
        float acc = 0.f;
        for (int s = 0; s < 20; ++s)
            acc += aS[(i * 4 + j) * 20 + s] * c3[((s * 7 + k) * 8 + e) * 16 + t];
        btS[((i * 4 + j) * 7 + k) * 16 + t] = acc;
    }
    __syncthreads();

    // step 3: Wt[c][row] = sum_t bt[i,j,k,t] * c4[t,l,f], row = ((i*4+j)*7+k)*4+l
    for (int row = threadIdx.x; row < 784; row += blockDim.x) {
        int l = row & 3;
        int k = (row >> 2) % 7;
        int j = (row / 28) & 3;
        int i = row / 112;
        float acc = 0.f;
        for (int t = 0; t < 16; ++t)
            acc += btS[((i * 4 + j) * 7 + k) * 16 + t] * c4[(t * 4 + l) * 4 + f];
        Wt[(size_t)c * WT_K + row] = (_Float16)acc;   // coalesced along row
    }
    if (threadIdx.x < WT_K - 784)
        Wt[(size_t)c * WT_K + 784 + threadIdx.x] = (_Float16)0.f;
}

// ---------------------------------------------------------------------------
// Kernel 2: fused GEMM (f16 WMMA, double-buffered LDS) + bias + ReLU + softmax.
// Block: 512 threads = 16 waves. M-tile = 32 rows, N = full 512 columns.
// ---------------------------------------------------------------------------
__global__ void __launch_bounds__(512)
tt_gemm_softmax(const float* __restrict__ x,       // [16384,784]
                const _Float16* __restrict__ Wt,   // [512][800]
                const float* __restrict__ bias,    // [512]
                float* __restrict__ out)           // [16384,512]
{
    extern __shared__ char smem[];
    _Float16* AsB  = (_Float16*)(smem + OFF_AS);
    _Float16* BsB  = (_Float16*)(smem + OFF_BS);
    float*    biasS = (float*)(smem + OFF_BIAS);
    float*    Cs   = (float*)(smem + OFF_CS);

    const int tid  = threadIdx.x;
    const int lane = tid & 31;
    const int w    = tid >> 5;          // wave 0..15
    const int mt   = w >> 3;            // 0..1  : 16-row sub-tile
    const int nbase = (w & 7) * 64;     // 64-col strip per wave
    const int rowBase = blockIdx.x * 32;

    biasS[tid] = bias[tid];             // blockDim == 512

    // --- staging thread maps ---
    // A: thread -> (row ar, k-pair akp); one float2 global load, one b32 LDS store
    const int ar  = tid >> 4;           // 0..31
    const int akp = (tid & 15) << 1;    // 0,2,..,30
    // B: thread -> (col bn + i*128, k-oct bk); four b128 loads/stores
    const int bk  = (tid & 3) * 8;      // 0,8,16,24
    const int bn  = tid >> 2;           // 0..127

    const float*    xp  = x  + (size_t)(rowBase + ar) * 784 + akp;
    const _Float16* wtp = Wt + (size_t)bn * WT_K + bk;

    const int arow = lane & 15;
    const int h    = lane >> 4;

    v8f acc0 = {}, acc1 = {}, acc2 = {}, acc3 = {};

    float2 av;
    v8h    bv0, bv1, bv2, bv3;

    auto stageLoad = [&](int kc) {
        const int K0 = kc * 32;
        if (kc < 24 || akp < 16) {
            av = *(const float2*)(xp + K0);
        } else {
            av.x = 0.f; av.y = 0.f;     // k >= 784 zero pad (last chunk only)
        }
        bv0 = *(const v8h*)(wtp + K0);
        bv1 = *(const v8h*)(wtp + (size_t)128 * WT_K + K0);
        bv2 = *(const v8h*)(wtp + (size_t)256 * WT_K + K0);
        bv3 = *(const v8h*)(wtp + (size_t)384 * WT_K + K0);
    };
    auto stageStore = [&](int buf) {
        _Float16* As = AsB + buf * (32 * AB_STRIDE);
        _Float16* Bs = BsB + buf * (512 * AB_STRIDE);
        v2h a2; a2[0] = (_Float16)av.x; a2[1] = (_Float16)av.y;
        *(v2h*)(As + ar * AB_STRIDE + akp) = a2;
        *(v8h*)(Bs + (bn      ) * AB_STRIDE + bk) = bv0;
        *(v8h*)(Bs + (bn + 128) * AB_STRIDE + bk) = bv1;
        *(v8h*)(Bs + (bn + 256) * AB_STRIDE + bk) = bv2;
        *(v8h*)(Bs + (bn + 384) * AB_STRIDE + bk) = bv3;
    };

    stageLoad(0);
    stageStore(0);

    for (int kc = 0; kc < 25; ++kc) {
        __syncthreads();
        const int buf = kc & 1;
        const bool more = (kc + 1 < 25);
        if (more) stageLoad(kc + 1);    // global loads in flight over compute

        _Float16* As = AsB + buf * (32 * AB_STRIDE);
        _Float16* Bs = BsB + buf * (512 * AB_STRIDE);

        // A fragment: row (mt*16+arow), K = 8h+{0..7} then 16+8h+{0..7}
        const _Float16* ap = As + (mt * 16 + arow) * AB_STRIDE + h * 8;
        v8h alo = *(const v8h*)(ap);
        v8h ahi = *(const v8h*)(ap + 16);
        v16h afrag = __builtin_shufflevector(alo, ahi,
                        0,1,2,3,4,5,6,7,8,9,10,11,12,13,14,15);

        #define BFRAG(NT, ACC)                                                   \
        {                                                                        \
            const _Float16* bp = Bs + (nbase + (NT)*16 + arow) * AB_STRIDE + h * 8; \
            v8h blo = *(const v8h*)(bp);                                         \
            v8h bhi = *(const v8h*)(bp + 16);                                    \
            v16h bfrag = __builtin_shufflevector(blo, bhi,                       \
                            0,1,2,3,4,5,6,7,8,9,10,11,12,13,14,15);              \
            ACC = __builtin_amdgcn_wmma_f32_16x16x32_f16(                        \
                      false, afrag, false, bfrag, (short)0, ACC, false, false);  \
        }
        BFRAG(0, acc0)
        BFRAG(1, acc1)
        BFRAG(2, acc2)
        BFRAG(3, acc3)
        #undef BFRAG

        if (more) stageStore(buf ^ 1);  // commit next chunk to other buffer
    }

    // epilogue: bias + ReLU into LDS. C/D layout: VGPR v -> M = v + 8*h, N = lane&15
    #pragma unroll
    for (int v = 0; v < 8; ++v) {
        int row = mt * 16 + v + 8 * h;
        int c0 = nbase + arow;
        Cs[row * C_STRIDE + c0]      = fmaxf(acc0[v] + biasS[c0],      0.f);
        Cs[row * C_STRIDE + c0 + 16] = fmaxf(acc1[v] + biasS[c0 + 16], 0.f);
        Cs[row * C_STRIDE + c0 + 32] = fmaxf(acc2[v] + biasS[c0 + 32], 0.f);
        Cs[row * C_STRIDE + c0 + 48] = fmaxf(acc3[v] + biasS[c0 + 48], 0.f);
    }
    __syncthreads();

    // softmax: 16 lanes per row, 32 rows
    {
        const int row = tid >> 4;      // 0..31
        const int sub = tid & 15;
        float* crow = Cs + row * C_STRIDE;

        float m = -3.402823466e+38f;
        #pragma unroll
        for (int j = 0; j < 32; ++j)
            m = fmaxf(m, crow[sub + j * 16]);
        #pragma unroll
        for (int d = 1; d < 16; d <<= 1)
            m = fmaxf(m, __shfl_xor(m, d, 32));

        float ssum = 0.f;
        #pragma unroll
        for (int j = 0; j < 32; ++j) {
            float ev = __expf(crow[sub + j * 16] - m);
            crow[sub + j * 16] = ev;
            ssum += ev;
        }
        #pragma unroll
        for (int d = 1; d < 16; d <<= 1)
            ssum += __shfl_xor(ssum, d, 32);
        float inv = 1.0f / ssum;

        float* orow = out + (size_t)(rowBase + row) * NOUT;
        #pragma unroll
        for (int j = 0; j < 32; ++j)
            orow[sub + j * 16] = crow[sub + j * 16] * inv;
    }
}

// ---------------------------------------------------------------------------
extern "C" void kernel_launch(void* const* d_in, const int* in_sizes, int n_in,
                              void* d_out, int out_size, void* d_ws, size_t ws_size,
                              hipStream_t stream) {
    const float* x     = (const float*)d_in[0];   // [16384,784]
    const float* core1 = (const float*)d_in[1];   // [7,4,20]
    const float* core2 = (const float*)d_in[2];   // [20,4,4,20]
    const float* core3 = (const float*)d_in[3];   // [20,7,8,16]
    const float* core4 = (const float*)d_in[4];   // [16,4,4]
    const float* bias  = (const float*)d_in[5];   // [512]
    float* out = (float*)d_out;

    _Float16* Wt = (_Float16*)d_ws;               // 512*800 f16 = 819,200 B

    tt_build_w<<<NOUT, 256, 0, stream>>>(core1, core2, core3, core4, Wt);

    tt_gemm_softmax<<<16384 / 32, 512, SMEM_TOTAL, stream>>>(x, Wt, bias, out);

    (void)in_sizes; (void)n_in; (void)out_size; (void)ws_size;
}
